// PFAE_PDN_68539088110347
// MI455X (gfx1250) — matrix operand order repaired and checked by reading, not compiled, not run on hardware
//
#include <hip/hip_runtime.h>
#include <hip/hip_bf16.h>
#include <math.h>

// ---------------------------------------------------------------------------
// PDNConv x10 GNN for MI455X (gfx1250, wave32, WMMA).
// Per layer:
//   1) k_lin_prep   : lin[cin,cout] f32 -> linT[cout_pad][K_pad] bf16 (B^T, padded)
//   2) k_edge_gate  : edge MLP -> sigmoid gate w[e]
//   3) deg/dis      : deg = 1 + sum w over dst ; dis = rsqrt(deg)
//   4) k_gemm_wmma  : xl = xbf @ lin (v_wmma_f32_16x16x32_bf16, double-buffered
//                     K loop) with fused epilogue out = xl * dis^2 (self loop)
//   5) k_scatter    : out[dst] += xl[src] * dis[src]*w*dis[dst] (hw f32 atomics)
//   6) k_bias_prelu : out = prelu(out + bias)
//   7) k_stats      : GraphNorm -> per-feature scale/shift (one-pass sum/sumsq)
//   8) k_apply_bf16 : next xbf = bf16(out*scale + shift), zero-padded
//  final layer: k_final : d_out = out + bias (f32)
// ---------------------------------------------------------------------------

typedef __attribute__((ext_vector_type(16))) __bf16        v16bf;
typedef __attribute__((ext_vector_type(8)))  float         f32x8;
typedef __attribute__((ext_vector_type(8)))  unsigned int  u32x8;

__device__ __forceinline__ float atomAddF(float* p, float v) {
  return __hip_atomic_fetch_add(p, v, __ATOMIC_RELAXED, __HIP_MEMORY_SCOPE_AGENT);
}

// ---------------- input conversion -----------------------------------------
__global__ void k_x_to_bf16(const float* __restrict__ x, __bf16* __restrict__ xbf, int total) {
  int id = blockIdx.x * blockDim.x + threadIdx.x;
  if (id < total) xbf[id] = (__bf16)x[id];
}

// linT[n*Kp + k] = (n<cout && k<cin) ? lin[k*cout + n] : 0
__global__ void k_lin_prep(const float* __restrict__ lin, __bf16* __restrict__ linT,
                           int cin, int cout, int Kp, int Cp) {
  int id = blockIdx.x * blockDim.x + threadIdx.x;
  if (id >= Cp * Kp) return;
  int n = id / Kp, k = id % Kp;
  float v = (n < cout && k < cin) ? lin[(size_t)k * cout + n] : 0.0f;
  linT[id] = (__bf16)v;
}

// ---------------- edge gate MLP ---------------------------------------------
__global__ void __launch_bounds__(256)
k_edge_gate(const float* __restrict__ ea, const float* __restrict__ w1,
            const float* __restrict__ b1, const float* __restrict__ w2,
            const float* __restrict__ b2, float* __restrict__ wout, int E) {
  __shared__ float sw1[8 * 32], sb1[32], sw2[32];
  sw1[threadIdx.x] = w1[threadIdx.x];
  if (threadIdx.x < 32) { sb1[threadIdx.x] = b1[threadIdx.x]; sw2[threadIdx.x] = w2[threadIdx.x]; }
  __syncthreads();
  int e = blockIdx.x * blockDim.x + threadIdx.x;
  if (e >= E) return;
  float a[8];
#pragma unroll
  for (int k = 0; k < 8; ++k) a[k] = ea[(size_t)e * 8 + k];
  float acc = b2[0];
#pragma unroll
  for (int j = 0; j < 32; ++j) {
    float h = sb1[j];
#pragma unroll
    for (int k = 0; k < 8; ++k) h = fmaf(a[k], sw1[k * 32 + j], h);
    h = fmaxf(h, 0.0f);
    acc = fmaf(h, sw2[j], acc);
  }
  wout[e] = 1.0f / (1.0f + __expf(-acc));
}

// ---------------- degree / normalization ------------------------------------
__global__ void k_deg_init(float* __restrict__ deg, int N) {
  int n = blockIdx.x * blockDim.x + threadIdx.x;
  if (n < N) deg[n] = 1.0f;  // self-loop weight
}
__global__ void k_deg_acc(const float* __restrict__ w, const int* __restrict__ dst,
                          float* __restrict__ deg, int E) {
  int e = blockIdx.x * blockDim.x + threadIdx.x;
  if (e < E) atomAddF(&deg[dst[e]], w[e]);
}
__global__ void k_dis(const float* __restrict__ deg, float* __restrict__ dis, int N) {
  int n = blockIdx.x * blockDim.x + threadIdx.x;
  if (n < N) {
    float d = deg[n];
    dis[n] = (d > 0.0f) ? rsqrtf(fmaxf(d, 1e-30f)) : 0.0f;
  }
}

// ---------------- WMMA GEMM -------------------------------------------------
// xl[N,Cp] = xbf[N,Kp] @ linT^T ; fused epilogue: out = xl * dis^2.
// One wave computes one 16x16 output tile; K loop is software double-buffered
// so each WMMA consumes fragments whose loads were issued one step earlier.
__device__ __forceinline__ v16bf load_frag(const __bf16* p_lo, const __bf16* p_hi) {
  const uint4 lo = *(const uint4*)p_lo;
  const uint4 hi = *(const uint4*)p_hi;
  u32x8 u;
  u[0] = lo.x; u[1] = lo.y; u[2] = lo.z; u[3] = lo.w;
  u[4] = hi.x; u[5] = hi.y; u[6] = hi.z; u[7] = hi.w;
  return __builtin_bit_cast(v16bf, u);
}

__global__ void __launch_bounds__(256)
k_gemm_wmma(const __bf16* __restrict__ xbf, const __bf16* __restrict__ linT,
            const float* __restrict__ dis, float* __restrict__ xl,
            float* __restrict__ outp, int n_rows, int Kp, int Cp) {
  const int lane = threadIdx.x & 31;
  const int wib  = threadIdx.x >> 5;
  const int tilesN = Cp >> 4;
  const int tilesM = n_rows >> 4;           // 30000 % 16 == 0
  int t = blockIdx.x * 8 + wib;
  if (t >= tilesM * tilesN) return;
  const int tileM = t / tilesN;             // waves in a block share tileM -> A reuse
  const int tileN = t - tileM * tilesN;
  const int half = lane >> 4;
  const int l16  = lane & 15;

  const __bf16* arow = xbf  + (size_t)(tileM * 16 + l16) * Kp;  // A row M = l16
  const __bf16* brow = linT + (size_t)(tileN * 16 + l16) * Kp;  // B col N = l16
  // A frag: elems 0..7 -> K = k0 + half*8 + (0..7); elems 8..15 -> +16.
  // B frag: same as A: elems 0..7 -> K = k0 + half*8 + (0..7); elems 8..15 -> +16.
  const __bf16* pa = arow + half * 8;
  const __bf16* pb = brow + half * 8;

  f32x8 c = {};
  v16bf a_cur = load_frag(pa, pa + 16);
  v16bf b_cur = load_frag(pb, pb + 16);
  for (int k0 = 32; k0 < Kp; k0 += 32) {
    __builtin_prefetch(arow + k0 + 32, 0, 3);   // near-cache prefetch of K+2 block
    __builtin_prefetch(brow + k0 + 32, 0, 3);
    v16bf a_nxt = load_frag(pa + k0, pa + k0 + 16);
    v16bf b_nxt = load_frag(pb + k0, pb + k0 + 16);
    c = __builtin_amdgcn_wmma_f32_16x16x32_bf16(
        false, a_cur, false, b_cur, (short)0, c, false, false);
    a_cur = a_nxt; b_cur = b_nxt;
  }
  c = __builtin_amdgcn_wmma_f32_16x16x32_bf16(
      false, a_cur, false, b_cur, (short)0, c, false, false);

  // C/D layout: VGPR r -> rows {r (lanes 0-15), 8+r (lanes 16-31)}, col = l16
  const int rowbase = tileM * 16 + half * 8;
  const int col     = tileN * 16 + l16;
  float* xrow = xl   + (size_t)rowbase * Cp + col;
  float* orow = outp + (size_t)rowbase * Cp + col;
#pragma unroll
  for (int r = 0; r < 8; ++r) {
    float d = dis[rowbase + r];
    xrow[(size_t)r * Cp] = c[r];
    orow[(size_t)r * Cp] = c[r] * d * d;    // self-loop contribution (norm = dis^2)
  }
}

// ---------------- edge scatter ------------------------------------------------
__global__ void __launch_bounds__(256)
k_scatter(const float* __restrict__ xl, const float* __restrict__ wgt,
          const float* __restrict__ dis, const int* __restrict__ src,
          const int* __restrict__ dst, float* __restrict__ out,
          int E, int cout, int stride) {
  int gwave  = (blockIdx.x * blockDim.x + threadIdx.x) >> 5;
  int lane   = threadIdx.x & 31;
  int nwaves = (gridDim.x * blockDim.x) >> 5;
  for (int e = gwave; e < E; e += nwaves) {
    int s = src[e], d = dst[e];
    float norm = dis[s] * wgt[e] * dis[d];
    const float* xr = xl + (size_t)s * stride;
    float* orow = out + (size_t)d * stride;
    for (int cpos = lane; cpos < cout; cpos += 32)
      atomAddF(&orow[cpos], xr[cpos] * norm);
  }
}

// ---------------- epilogues ---------------------------------------------------
__global__ void k_bias_prelu(float* __restrict__ h, const float* __restrict__ bias,
                             const float* __restrict__ prelu, int N, int cout, int stride) {
  int id = blockIdx.x * blockDim.x + threadIdx.x;
  if (id >= N * cout) return;
  int n = id / cout, cpos = id - n * cout;
  size_t o = (size_t)n * stride + cpos;
  float v = h[o] + bias[cpos];
  float a = prelu[0];
  h[o] = (v >= 0.0f) ? v : a * v;
}

// GraphNorm: one-pass sum/sumsq per feature -> per-feature scale/shift
__global__ void __launch_bounds__(256)
k_stats(const float* __restrict__ h, int N, int stride,
        const float* __restrict__ gnw, const float* __restrict__ gnb,
        const float* __restrict__ gnms, float* __restrict__ scl, float* __restrict__ shf) {
  int cpos = blockIdx.x;
  __shared__ float ssum[256], ssq[256];
  float s = 0.0f, q = 0.0f;
  for (int n = threadIdx.x; n < N; n += 256) {
    float v = h[(size_t)n * stride + cpos];
    s += v; q += v * v;
  }
  ssum[threadIdx.x] = s; ssq[threadIdx.x] = q;
  __syncthreads();
  for (int ofs = 128; ofs > 0; ofs >>= 1) {
    if (threadIdx.x < ofs) {
      ssum[threadIdx.x] += ssum[threadIdx.x + ofs];
      ssq[threadIdx.x]  += ssq[threadIdx.x + ofs];
    }
    __syncthreads();
  }
  if (threadIdx.x == 0) {
    float inv_n  = 1.0f / (float)N;
    float mean   = ssum[0] * inv_n;
    float meansq = ssq[0] * inv_n;
    float a   = mean * gnms[cpos];                // shift applied to x
    float var = meansq - 2.0f * a * mean + a * a; // var of (x - a)
    float is  = rsqrtf(var + 1e-5f);
    float sc  = gnw[cpos] * is;
    scl[cpos] = sc;
    shf[cpos] = gnb[cpos] - a * sc;
  }
}

__global__ void k_apply_bf16(const float* __restrict__ h, int stride,
                             const float* __restrict__ scl, const float* __restrict__ shf,
                             __bf16* __restrict__ xbf, int N, int cout, int Kn) {
  int id = blockIdx.x * blockDim.x + threadIdx.x;
  if (id >= N * Kn) return;
  int n = id / Kn, cpos = id - n * Kn;
  float v = 0.0f;
  if (cpos < cout) v = h[(size_t)n * stride + cpos] * scl[cpos] + shf[cpos];
  xbf[id] = (__bf16)v;
}

__global__ void k_final(const float* __restrict__ h, const float* __restrict__ bias,
                        float* __restrict__ out, int N, int cout, int stride) {
  int id = blockIdx.x * blockDim.x + threadIdx.x;
  if (id >= N * cout) return;
  int n = id / cout, cpos = id - n * cout;
  out[id] = h[(size_t)n * stride + cpos] + bias[cpos];
}

// ---------------------------------------------------------------------------
extern "C" void kernel_launch(void* const* d_in, const int* in_sizes, int n_in,
                              void* d_out, int out_size, void* d_ws, size_t ws_size,
                              hipStream_t stream) {
  (void)in_sizes; (void)n_in; (void)out_size; (void)ws_size;
  static const int DIMS[11] = {256, 237, 218, 199, 180, 160, 141, 122, 103, 84, 64};
  const int N = 30000, E = 480000;

  const float* x  = (const float*)d_in[0];
  const int*   ei = (const int*)d_in[1];
  const float* ea = (const float*)d_in[2];
  const int* srcp = ei;       // edge_index[0]
  const int* dstp = ei + E;   // edge_index[1]

  // params: JAX pytree flatten = list order, dict keys sorted alphabetically.
  // i<9: b1,b2,bias,gn_bias,gn_mean_scale,gn_weight,lin,prelu,w1,w2
  // i=9: b1,b2,bias,lin,w1,w2
  struct Layer { const float *b1,*b2,*bias,*gnb,*gnms,*gnw,*lin,*prelu,*w1,*w2; };
  Layer L[10];
  int idx = 3;
  for (int i = 0; i < 10; ++i) {
    L[i].b1   = (const float*)d_in[idx++];
    L[i].b2   = (const float*)d_in[idx++];
    L[i].bias = (const float*)d_in[idx++];
    L[i].gnb = L[i].gnms = L[i].gnw = L[i].prelu = nullptr;
    if (i < 9) {
      L[i].gnb  = (const float*)d_in[idx++];
      L[i].gnms = (const float*)d_in[idx++];
      L[i].gnw  = (const float*)d_in[idx++];
    }
    L[i].lin = (const float*)d_in[idx++];
    if (i < 9) L[i].prelu = (const float*)d_in[idx++];
    L[i].w1 = (const float*)d_in[idx++];
    L[i].w2 = (const float*)d_in[idx++];
  }

  // workspace carve (256B aligned)
  char* base = (char*)d_ws; size_t off = 0;
  auto carve = [&](size_t bytes) -> void* {
    void* q = base + off;
    off += (bytes + 255) & ~(size_t)255;
    return q;
  };
  __bf16* xbf  = (__bf16*)carve((size_t)N * 256 * sizeof(__bf16));
  float*  xl   = (float*) carve((size_t)N * 256 * sizeof(float));
  float*  outb = (float*) carve((size_t)N * 256 * sizeof(float));
  float*  wg   = (float*) carve((size_t)E * sizeof(float));
  float*  deg  = (float*) carve((size_t)N * sizeof(float));
  float*  dis  = (float*) carve((size_t)N * sizeof(float));
  __bf16* linT = (__bf16*)carve((size_t)256 * 256 * sizeof(__bf16));
  float*  scl  = (float*) carve(256 * sizeof(float));
  float*  shf  = (float*) carve(256 * sizeof(float));

  auto pad32 = [](int v) { return (v + 31) & ~31; };
  const int B = 256;

  // initial bf16 input (cin = 256, already 32-aligned)
  k_x_to_bf16<<<(N * 256 + B - 1) / B, B, 0, stream>>>(x, xbf, N * 256);

  for (int i = 0; i < 10; ++i) {
    const int cin = DIMS[i], cout = DIMS[i + 1];
    const int Kp = pad32(cin);     // xbf row stride (== prev layer's Cp)
    const int Cp = pad32(cout);    // xl/out row stride; also next layer's Kp

    k_lin_prep<<<(Cp * Kp + B - 1) / B, B, 0, stream>>>(L[i].lin, linT, cin, cout, Kp, Cp);
    k_edge_gate<<<(E + B - 1) / B, B, 0, stream>>>(ea, L[i].w1, L[i].b1, L[i].w2, L[i].b2, wg, E);
    k_deg_init<<<(N + B - 1) / B, B, 0, stream>>>(deg, N);
    k_deg_acc<<<(E + B - 1) / B, B, 0, stream>>>(wg, dstp, deg, E);
    k_dis<<<(N + B - 1) / B, B, 0, stream>>>(deg, dis, N);

    const int tiles = (N / 16) * (Cp / 16);
    k_gemm_wmma<<<(tiles + 7) / 8, B, 0, stream>>>(xbf, linT, dis, xl, outb, N, Kp, Cp);

    k_scatter<<<2048, B, 0, stream>>>(xl, wg, dis, srcp, dstp, outb, E, cout, Cp);

    if (i < 9) {
      k_bias_prelu<<<(N * cout + B - 1) / B, B, 0, stream>>>(outb, L[i].bias, L[i].prelu, N, cout, Cp);
      k_stats<<<cout, B, 0, stream>>>(outb, N, Cp, L[i].gnw, L[i].gnb, L[i].gnms, scl, shf);
      const int Kn = pad32(cout);  // next layer input width (== Cp)
      k_apply_bf16<<<(N * Kn + B - 1) / B, B, 0, stream>>>(outb, Cp, scl, shf, xbf, N, cout, Kn);
    } else {
      k_final<<<(N * cout + B - 1) / B, B, 0, stream>>>(outb, L[i].bias, (float*)d_out, N, cout, Cp);
    }
  }
}
